// MessagePassingNetwork_62526133895719
// MI455X (gfx1250) — matrix operand order repaired and checked by reading, not compiled, not run on hardware
//
#include <hip/hip_runtime.h>
#include <hip/hip_bf16.h>

#define DEV __device__ __forceinline__

typedef __attribute__((ext_vector_type(16))) __bf16 bf16x16;
typedef __attribute__((ext_vector_type(8)))  __bf16 bf16x8;
typedef __attribute__((ext_vector_type(8)))  float  f32x8;

static constexpr int D = 128;
static constexpr int M = 32;
static constexpr int L = 3;

// packed bf16 weight offsets within one layer (elements)
static constexpr int PK_WS    = 0;        // 128x32
static constexpr int PK_WR    = 4096;     // 128x32
static constexpr int PK_WE    = 8192;     // 32x32
static constexpr int PK_WOUT  = 9216;     // 32x128
static constexpr int PK_WG1V  = 13312;    // 128x128
static constexpr int PK_WG1G  = 29696;    // 128x128
static constexpr int PK_WG2V  = 46080;    // 128x128
static constexpr int PK_WG2G  = 62464;    // 128x128
static constexpr int PK_WF    = 78848;    // 128x128 (layer chunk of W_final)
static constexpr int PK_LAYER = 95232;

DEV float silu_f(float x){ return x / (1.f + __expf(-x)); }

DEV f32x8 wmma_bf16(bf16x16 a, bf16x16 b, f32x8 c){
  return __builtin_amdgcn_wmma_f32_16x16x32_bf16(false, a, false, b, (short)0, c, false, false);
}

// A-fragment (16x32 bf16) from LDS row pointer (row r already applied):
// lane (r, h): elems 0..7 = A[r][8h+j], elems 8..15 = A[r][16+8h+j]
DEV bf16x16 load_a_lds(const __bf16* rowp, int h){
  union { bf16x16 v; bf16x8 p[2]; } u;
  u.p[0] = *(const bf16x8*)(rowp + 8*h);
  u.p[1] = *(const bf16x8*)(rowp + 16 + 8*h);
  return u.v;
}

// Same A-fragment but sourced from global f32 with on-the-fly bf16 convert
DEV bf16x16 load_a_f32(const float* rowp, int h){
  bf16x16 a;
  const float* p0 = rowp + 8*h;
  const float* p1 = rowp + 16 + 8*h;
#pragma unroll
  for (int j=0;j<8;++j){ a[j] = (__bf16)p0[j]; a[8+j] = (__bf16)p1[j]; }
  return a;
}

// B-fragment (32x16 bf16) from pre-swizzled packed weights: 32B contiguous per lane
DEV bf16x16 load_b_pk(const __bf16* pk, int tile, int lane){
  return *(const bf16x16*)(pk + ((size_t)(tile*32 + lane) << 4));
}

// ---------------------------------------------------------------------------
// Weight packing: dst[tile*512 + lane*16 + j] = W[32kt + (j<8 ? 8(lane/16) + j : 16 + 8(lane/16) + j-8)][16nt + lane%16]
// blockIdx.y = matrix id (0..8), blockIdx.z = layer
__global__ void __launch_bounds__(256)
pack_weights_kernel(const float* Ws, const float* Wr, const float* We,
                    const float* Wout, const float* Wg1v, const float* Wg1g,
                    const float* Wg2v, const float* Wg2g, const float* Wf,
                    __bf16* pk)
{
  const int l = blockIdx.z;
  const int m = blockIdx.y;
  const int Ktab[9]  = {128,128, 32, 32,128,128,128,128,128};
  const int Ntab[9]  = { 32, 32, 32,128,128,128,128,128,128};
  const int Otab[9]  = {PK_WS,PK_WR,PK_WE,PK_WOUT,PK_WG1V,PK_WG1G,PK_WG2V,PK_WG2G,PK_WF};
  const float* src;
  switch (m){
    case 0: src = Ws   + (size_t)l*128*32;  break;
    case 1: src = Wr   + (size_t)l*128*32;  break;
    case 2: src = We   + (size_t)l*32*32;   break;
    case 3: src = Wout + (size_t)l*32*128;  break;
    case 4: src = Wg1v + (size_t)l*128*128; break;
    case 5: src = Wg1g + (size_t)l*128*128; break;
    case 6: src = Wg2v + (size_t)l*128*128; break;
    case 7: src = Wg2g + (size_t)l*128*128; break;
    default: src = Wf  + (size_t)l*128*128; break;  // W_final rows l*128..
  }
  const int K = Ktab[m], Nc = Ntab[m];
  const int idx = blockIdx.x*blockDim.x + threadIdx.x;
  if (idx >= K*Nc) return;
  const int j    = idx & 15;
  const int lane = (idx >> 4) & 31;
  const int tile = idx >> 9;
  const int NT = Nc >> 4;
  const int kt = tile / NT, nt = tile - kt*NT;
  const int srow = 32*kt + (j < 8 ? 8*(lane >> 4) + j : 16 + 8*(lane >> 4) + (j - 8));
  const int scol = 16*nt + (lane & 15);
  pk[(size_t)l*PK_LAYER + Otab[m] + idx] = (__bf16)src[(size_t)srow*Nc + scol];
}

// ---------------------------------------------------------------------------
__global__ void __launch_bounds__(256)
scatter_norm_kernel(const float* __restrict__ e_norm, const int* __restrict__ rcv,
                    float* __restrict__ norm, int E)
{
  int i = blockIdx.x*blockDim.x + threadIdx.x;
  if (i < E) atomicAdd(&norm[rcv[i]], e_norm[i]);
}

__global__ void __launch_bounds__(256)
finalize_norm_kernel(float* __restrict__ norm, int N)
{
  int i = blockIdx.x*blockDim.x + threadIdx.x;
  if (i < N) norm[i] = 1.f / (norm[i] + 1.f);
}

// ---------------------------------------------------------------------------
// Out[N,32] = X[N,128] @ Wpk(128x32, packed) (+ bias). One wave per 16-row tile.
__global__ void __launch_bounds__(256)
proj_kernel(const float* __restrict__ X, const __bf16* __restrict__ Wpk,
            const float* __restrict__ bias, float* __restrict__ Out, int ntiles)
{
  const int lane = threadIdx.x & 31;
  const int wave = threadIdx.x >> 5;
  const int tile = blockIdx.x*8 + wave;
  if (tile >= ntiles) return;          // wave-uniform; no barriers in this kernel
  const int n = lane & 15, h = lane >> 4;
  const int n0 = tile * 16;
  f32x8 acc0 = {}, acc1 = {};
#pragma unroll
  for (int ck = 0; ck < 4; ++ck){
    bf16x16 a = load_a_f32(X + (size_t)(n0 + n)*128 + 32*ck, h);
    acc0 = wmma_bf16(a, load_b_pk(Wpk, ck*2 + 0, lane), acc0);
    acc1 = wmma_bf16(a, load_b_pk(Wpk, ck*2 + 1, lane), acc1);
  }
  const float b0 = bias ? bias[n]      : 0.f;
  const float b1 = bias ? bias[16 + n] : 0.f;
#pragma unroll
  for (int v = 0; v < 8; ++v){
    const size_t row = (size_t)(n0 + v + 8*h);
    Out[row*32 + n]      = acc0[v] + b0;
    Out[row*32 + 16 + n] = acc1[v] + b1;
  }
}

// ---------------------------------------------------------------------------
// Per 16-edge tile (one wave): x = HS[snd]+HR[rcv]; g = silu(LN(x));
// m = g * (e_embed @ W_e); atomic scatter-add into msg[rcv].
__global__ void __launch_bounds__(256)
edge_kernel(const float* __restrict__ HS, const float* __restrict__ HR,
            const float* __restrict__ Ee, const int* __restrict__ snd,
            const int* __restrict__ rcv, const __bf16* __restrict__ Wepk,
            const float* __restrict__ lnS, const float* __restrict__ lnB,
            float* __restrict__ msg, int E)
{
  __shared__ float xsh[8][16][32];
  __shared__ float mush[8][16];
  __shared__ float rssh[8][16];
  __shared__ int   dsh[8][16];
  const int lane = threadIdx.x & 31;
  const int wave = threadIdx.x >> 5;
  const int n = lane & 15, h = lane >> 4;
  const long long e0 = ((long long)blockIdx.x*8 + wave) * 16;
  const long long er = e0 + n;
  const long long erow = er < E ? er : (long long)E - 1;   // clamp (no early return: barrier below)
  const int sidx = snd[erow];
  const int ridx = rcv[erow];
  if (h == 0) dsh[wave][n] = ridx;
  const float* hsp = HS + (size_t)sidx*32 + 16*h;
  const float* hrp = HR + (size_t)ridx*32 + 16*h;
  float sum = 0.f, sq = 0.f;
#pragma unroll
  for (int j = 0; j < 16; ++j){
    float x = hsp[j] + hrp[j];
    xsh[wave][n][16*h + j] = x;
    sum += x; sq += x*x;
  }
  sum += __shfl_xor(sum, 16);
  sq  += __shfl_xor(sq, 16);
  const float mu = sum * (1.f/32.f);
  const float rs = rsqrtf(sq*(1.f/32.f) - mu*mu + 1e-6f);
  if (h == 0){ mush[wave][n] = mu; rssh[wave][n] = rs; }

  bf16x16 a = load_a_f32(Ee + (size_t)erow*32, h);     // A rows are this tile's edges
  f32x8 acc0 = {}, acc1 = {};
  acc0 = wmma_bf16(a, load_b_pk(Wepk, 0, lane), acc0);
  acc1 = wmma_bf16(a, load_b_pk(Wepk, 1, lane), acc1);
  const float s0 = lnS[n],      b0 = lnB[n];
  const float s1 = lnS[16 + n], b1 = lnB[16 + n];
  __syncthreads();
#pragma unroll
  for (int v = 0; v < 8; ++v){
    const int mr = v + 8*h;
    if (e0 + mr < E){
      const float mu2 = mush[wave][mr], rs2 = rssh[wave][mr];
      const float g0 = silu_f((xsh[wave][mr][n]      - mu2)*rs2*s0 + b0);
      const float g1 = silu_f((xsh[wave][mr][16 + n] - mu2)*rs2*s1 + b1);
      float* dst = msg + (size_t)dsh[wave][mr]*32;
      atomicAdd(dst + n,      g0*acc0[v]);
      atomicAdd(dst + 16 + n, g1*acc1[v]);
    }
  }
}

// ---------------------------------------------------------------------------
// Per 16-node tile (256 threads, 8 waves = 8 col-tiles of 128):
// m2 = silu((msg*norm)@W_out); y = glu1(r)+m2; r' = glu2(y)+r; out += r'@Wf_l
__global__ void __launch_bounds__(256)
node_kernel(float* __restrict__ r, const float* __restrict__ msg,
            const float* __restrict__ norm, const __bf16* __restrict__ pkl,
            const float* __restrict__ bg1v, const float* __restrict__ bg2v,
            const float* __restrict__ b_final, float* __restrict__ out,
            int first_layer)
{
  __shared__ __align__(16) __bf16 rT [16*128];
  __shared__ float                 rF [16*128];
  __shared__ __align__(16) __bf16 mT [16*32];
  __shared__ __align__(16) __bf16 yT [16*128];
  __shared__ __align__(16) __bf16 rnT[16*128];

  const int t = threadIdx.x;
  const int lane = t & 31;
  const int ct = t >> 5;               // wave id == output column tile
  const int n = lane & 15, h = lane >> 4;
  const int node0 = blockIdx.x * 16;

  for (int i = t; i < 16*128; i += 256){
    float v = r[(size_t)node0*128 + i];
    rF[i] = v; rT[i] = (__bf16)v;
  }
  for (int i = t; i < 16*32; i += 256){
    mT[i] = (__bf16)(msg[(size_t)node0*32 + i] * norm[node0 + (i >> 5)]);
  }
  __syncthreads();

  // message transform
  f32x8 accm = {};
  accm = wmma_bf16(load_a_lds(mT + n*32, h), load_b_pk(pkl + PK_WOUT, ct, lane), accm);

  // GLU1 on r
  f32x8 accv = {}, accg = {};
  {
    const float bv = bg1v[16*ct + n];
#pragma unroll
    for (int k = 0; k < 8; ++k) accv[k] = bv;
#pragma unroll
    for (int ck = 0; ck < 4; ++ck){
      bf16x16 a = load_a_lds(rT + n*128 + 32*ck, h);
      accv = wmma_bf16(a, load_b_pk(pkl + PK_WG1V, ck*8 + ct, lane), accv);
      accg = wmma_bf16(a, load_b_pk(pkl + PK_WG1G, ck*8 + ct, lane), accg);
    }
  }
#pragma unroll
  for (int k = 0; k < 8; ++k){
    float y = accv[k]*silu_f(accg[k]) + silu_f(accm[k]);
    yT[(k + 8*h)*128 + 16*ct + n] = (__bf16)y;
  }
  __syncthreads();

  // GLU2 on y + residual
  f32x8 a2v = {}, a2g = {};
  {
    const float bv = bg2v[16*ct + n];
#pragma unroll
    for (int k = 0; k < 8; ++k) a2v[k] = bv;
#pragma unroll
    for (int ck = 0; ck < 4; ++ck){
      bf16x16 a = load_a_lds(yT + n*128 + 32*ck, h);
      a2v = wmma_bf16(a, load_b_pk(pkl + PK_WG2V, ck*8 + ct, lane), a2v);
      a2g = wmma_bf16(a, load_b_pk(pkl + PK_WG2G, ck*8 + ct, lane), a2g);
    }
  }
#pragma unroll
  for (int k = 0; k < 8; ++k){
    const int row = k + 8*h;
    float val = a2v[k]*silu_f(a2g[k]) + rF[row*128 + 16*ct + n];
    r[(size_t)(node0 + row)*128 + 16*ct + n] = val;    // in-place: block owns rows
    rnT[row*128 + 16*ct + n] = (__bf16)val;
  }
  __syncthreads();

  // out accumulation: this layer's W_final chunk
  f32x8 acco = {};
#pragma unroll
  for (int ck = 0; ck < 4; ++ck){
    acco = wmma_bf16(load_a_lds(rnT + n*128 + 32*ck, h),
                     load_b_pk(pkl + PK_WF, ck*8 + ct, lane), acco);
  }
#pragma unroll
  for (int k = 0; k < 8; ++k){
    const size_t oi = (size_t)(node0 + k + 8*h)*128 + 16*ct + n;
    const float base = first_layer ? b_final[16*ct + n] : out[oi];
    out[oi] = base + acco[k];
  }
}

// ---------------------------------------------------------------------------
extern "C" void kernel_launch(void* const* d_in, const int* in_sizes, int n_in,
                              void* d_out, int out_size, void* d_ws, size_t ws_size,
                              hipStream_t stream)
{
  const float* s_embed = (const float*)d_in[0];
  const float* r_embed = (const float*)d_in[1];
  const float* e_embed = (const float*)d_in[2];
  const float* e_norm  = (const float*)d_in[3];
  const int*   senders = (const int*)d_in[4];
  const int*   recvs   = (const int*)d_in[5];
  const float* W_s  = (const float*)d_in[6];
  const float* b_s  = (const float*)d_in[7];
  const float* W_r  = (const float*)d_in[8];
  const float* lnS  = (const float*)d_in[9];
  const float* lnB  = (const float*)d_in[10];
  const float* W_e  = (const float*)d_in[11];
  const float* W_out= (const float*)d_in[12];
  const float* Wg1v = (const float*)d_in[13];
  const float* bg1v = (const float*)d_in[14];
  const float* Wg1g = (const float*)d_in[15];
  const float* Wg2v = (const float*)d_in[16];
  const float* bg2v = (const float*)d_in[17];
  const float* Wg2g = (const float*)d_in[18];
  const float* W_f  = (const float*)d_in[19];
  const float* b_f  = (const float*)d_in[20];
  (void)n_in; (void)out_size;

  const int N = in_sizes[0] / D;       // 100000 (multiple of 16)
  const int E = in_sizes[3];           // 1600000

  auto align256 = [](size_t x){ return (x + 255) & ~(size_t)255; };
  char* ws = (char*)d_ws;
  size_t off = 0;
  float*  r_ws = (float*)(ws + off);  off = align256(off + (size_t)N*D*4);
  float*  HS   = (float*)(ws + off);  off = align256(off + (size_t)N*M*4);
  float*  HR   = (float*)(ws + off);  off = align256(off + (size_t)N*M*4);
  float*  msg  = (float*)(ws + off);  off = align256(off + (size_t)N*M*4);
  float*  norm = (float*)(ws + off);  off = align256(off + (size_t)N*4);
  __bf16* pk   = (__bf16*)(ws + off); off = align256(off + (size_t)L*PK_LAYER*2);
  (void)ws_size;  // total ~91 MB

  pack_weights_kernel<<<dim3(64, 9, L), 256, 0, stream>>>(
      W_s, W_r, W_e, W_out, Wg1v, Wg1g, Wg2v, Wg2g, W_f, pk);

  hipMemsetAsync(norm, 0, (size_t)N*4, stream);
  scatter_norm_kernel<<<(E + 255)/256, 256, 0, stream>>>(e_norm, recvs, norm, E);
  finalize_norm_kernel<<<(N + 255)/256, 256, 0, stream>>>(norm, N);
  hipMemcpyAsync(r_ws, r_embed, (size_t)N*D*4, hipMemcpyDeviceToDevice, stream);

  const int ntiles   = N / 16;
  const int projGrid = (ntiles + 7) / 8;
  const int etiles   = (E + 15) / 16;
  const int edgeGrid = (etiles + 7) / 8;

  for (int l = 0; l < L; ++l){
    const __bf16* pkl = pk + (size_t)l*PK_LAYER;
    proj_kernel<<<projGrid, 256, 0, stream>>>(s_embed, pkl + PK_WS, b_s + l*M, HS, ntiles);
    proj_kernel<<<projGrid, 256, 0, stream>>>(r_ws,    pkl + PK_WR, nullptr,   HR, ntiles);
    hipMemsetAsync(msg, 0, (size_t)N*M*4, stream);
    edge_kernel<<<edgeGrid, 256, 0, stream>>>(HS, HR, e_embed, senders, recvs,
                                              pkl + PK_WE, lnS + l*M, lnB + l*M, msg, E);
    node_kernel<<<ntiles, 256, 0, stream>>>(r_ws, msg, norm, pkl,
                                            bg1v + l*D, bg2v + l*D, b_f,
                                            (float*)d_out, l == 0);
  }
}
